// EncoderBlock_85066122265673
// MI455X (gfx1250) — matrix operand order repaired and checked by reading, not compiled, not run on hardware
//
#include <hip/hip_runtime.h>
#include <hip/hip_bf16.h>
#include <math.h>

#define B_   2
#define S_   2048
#define D_   1024
#define H_   16
#define HD_  64
#define FFN_ 4096
#define MSZ  (B_ * S_)   // 4096 rows

typedef __attribute__((ext_vector_type(16))) __bf16 bf16x16;
typedef __attribute__((ext_vector_type(8)))  __bf16 bf16x8;
typedef __attribute__((ext_vector_type(8)))  float  f32x8;
typedef unsigned int u32x4 __attribute__((ext_vector_type(4)));
typedef int          i32x8 __attribute__((ext_vector_type(8)));
typedef int          i32x4 __attribute__((ext_vector_type(4)));

union frag_u { struct { bf16x8 lo, hi; } p; bf16x16 v; };

static __device__ __forceinline__ bf16x16 make_frag(bf16x8 lo, bf16x8 hi) {
    frag_u u; u.p.lo = lo; u.p.hi = hi; return u.v;
}
static __device__ __forceinline__ f32x8 zero8() {
    f32x8 z = {0.f,0.f,0.f,0.f,0.f,0.f,0.f,0.f}; return z;
}

// ---------------------------------------------------------------------------
// TDM: DMA one 128x32 bf16 tile (row stride = stride_elems) into LDS with
// hardware padding: 16 DWORDs (64B row) then +4 DWORDs pad -> 80B LDS stride.
// D# layout per CDNA5 ISA 8.3/8.4: group0 = {flags, lds_addr, global_addr,
// type=2}; group1 = {wg_mask=0, data_size=2B, pad cfg, dims, tile, strides}.
// Toolchain uses the 6-arg builtin: (g0, g1, g2, g3, g_extra, cpol).
// ---------------------------------------------------------------------------
static __device__ __forceinline__ void tdm_load_tile(
    unsigned lds_off, const __bf16* gptr, unsigned stride_elems)
{
    unsigned long long ga = (unsigned long long)(size_t)gptr;
    u32x4 g0;
    g0[0] = 1u;                                             // count=1 (valid)
    g0[1] = lds_off;                                        // LDS byte addr
    g0[2] = (unsigned)(ga & 0xFFFFFFFFu);                   // global addr lo
    g0[3] = (unsigned)((ga >> 32) & 0x01FFFFFFu) | (2u << 30); // addr hi|type=2
    unsigned td0 = stride_elems;        // tensor_dim0 (never OOB for us)
    unsigned td1 = 1u << 20;            // tensor_dim1 large
    i32x8 g1;
    g1[0] = (int)((1u << 16)            // data_size = 1 -> 2 bytes
                | (1u << 20)            // pad_enable
                | (3u << 22)            // pad_interval = 16 DWORDs
                | (3u << 25));          // pad_amount   = 4 DWORDs
    g1[1] = (int)((td0 & 0xFFFFu) << 16);                   // dim0[15:0]
    g1[2] = (int)((td0 >> 16) | ((td1 & 0xFFFFu) << 16));   // dim0[31:16]|dim1
    g1[3] = (int)((td1 >> 16) | (32u << 16));               // dim1 hi | tile_dim0=32
    g1[4] = (int)(128u);                                    // tile_dim1=128, dim2=0
    g1[5] = (int)(stride_elems);                            // dim0_stride lo
    g1[6] = 0;                                              // stride hi | dim1_stride lo
    g1[7] = 0;
    i32x4 z4 = {0, 0, 0, 0};                                // 2-D: groups 2/3 null
    i32x8 z8 = {0, 0, 0, 0, 0, 0, 0, 0};
    __builtin_amdgcn_tensor_load_to_lds(g0, g1, z4, z4, z8, 0);
}

// ---------------------------------------------------------------------------
// Conversion / weight-transpose kernels (one-time per call; all bf16 staging)
// ---------------------------------------------------------------------------
__global__ void cvt_f32_bf16(const float* __restrict__ in, __bf16* __restrict__ out, int n) {
    int i = blockIdx.x * blockDim.x + threadIdx.x;
    if (i < n) out[i] = (__bf16)in[i];
}

// out[n*K + k] = in[k*N + n]   (plain weight [K][N] -> Bt [N][K])
__global__ void transpose_wt(const float* __restrict__ in, __bf16* __restrict__ out, int K, int N) {
    int idx = blockIdx.x * blockDim.x + threadIdx.x;
    if (idx < K * N) {
        int n = idx / K, k = idx % K;
        out[idx] = (__bf16)in[(size_t)k * N + n];
    }
}

// wq/wk/wv: (H, D, HD) with n = h*64+e, k = d -> Bt [1024][1024]
__global__ void transpose_wqkv(const float* __restrict__ in, __bf16* __restrict__ out) {
    int idx = blockIdx.x * blockDim.x + threadIdx.x;
    if (idx < H_ * D_ * HD_) {
        int n = idx >> 10, k = idx & 1023;
        int h = n >> 6,  e = n & 63;
        out[idx] = (__bf16)in[((size_t)h * D_ + k) * HD_ + e];
    }
}

// ---------------------------------------------------------------------------
// Tiled WMMA GEMM: C[M][N] = A[M][K] * Bt[N][K]^T  (bf16 in, fp32 accum)
// Block 256 thr = 8 waves; tile 128x128, K-step 32, TDM double-buffered LDS.
// Wave0 issues A-tile DMA, wave1 issues B-tile DMA (TENSORcnt, in-order).
// MODE 0: Q/K scatter [B,H,S,HD] bf16 (+bias)
// MODE 1: V scatter transposed [B,H,HD,S] bf16 (+bias)
// MODE 2: fp32 out = val + bias + resid  (proj-residual / FFN2-residual)
// MODE 3: bf16 out = relu(val + bias)    (FFN1)
// ---------------------------------------------------------------------------
#define LDSW 40   // padded LDS row stride in bf16 elems (64B data + 16B pad)

template<int MODE>
__global__ __launch_bounds__(256) void gemm_bf16(
    const __bf16* __restrict__ A, const __bf16* __restrict__ Bt,
    const float* __restrict__ bias, const float* __restrict__ resid,
    void* __restrict__ out, int M, int N, int K)
{
    __shared__ __align__(16) __bf16 lA[2][128 * LDSW];
    __shared__ __align__(16) __bf16 lB[2][128 * LDSW];

    const int tid  = threadIdx.x;
    const int lane = tid & 31;
    const int wave = tid >> 5;
    const int wm   = wave & 1;      // 0..1 -> 64-row half
    const int wn   = wave >> 1;     // 0..3 -> 32-col quarter
    const int bm   = blockIdx.y * 128;
    const int bn   = blockIdx.x * 128;
    const int half = lane >> 4;
    const int l16  = lane & 15;

    f32x8 acc[4][2];
    for (int mt = 0; mt < 4; ++mt)
        for (int nt = 0; nt < 2; ++nt) acc[mt][nt] = zero8();

    const __bf16* gA = A  + (size_t)bm * K;
    const __bf16* gB = Bt + (size_t)bn * K;
    const unsigned offA[2] = { (unsigned)(size_t)(void*)&lA[0][0],
                               (unsigned)(size_t)(void*)&lA[1][0] };
    const unsigned offB[2] = { (unsigned)(size_t)(void*)&lB[0][0],
                               (unsigned)(size_t)(void*)&lB[1][0] };

    const int nT = K >> 5;
    // prologue: DMA tile 0
    if (wave == 0)      tdm_load_tile(offA[0], gA, (unsigned)K);
    else if (wave == 1) tdm_load_tile(offB[0], gB, (unsigned)K);

    for (int t = 0; t < nT; ++t) {
        if (wave < 2) {
            if (t + 1 < nT) {
                // prefetch tile t+1, then wait for tile t (in-order per wave)
                if (wave == 0) tdm_load_tile(offA[(t + 1) & 1], gA + (t + 1) * 32, (unsigned)K);
                else           tdm_load_tile(offB[(t + 1) & 1], gB + (t + 1) * 32, (unsigned)K);
                __builtin_amdgcn_s_wait_tensorcnt(1);
            } else {
                __builtin_amdgcn_s_wait_tensorcnt(0);
            }
        }
        __syncthreads();
        asm volatile("" ::: "memory");   // LDS contents written by TDM

        const __bf16* cA = &lA[t & 1][0];
        const __bf16* cB = &lB[t & 1][0];

        bf16x16 afr[4], bfr[2];
        #pragma unroll
        for (int mt = 0; mt < 4; ++mt) {
            const __bf16* r = cA + (wm * 64 + mt * 16 + l16) * LDSW;
            afr[mt] = make_frag(*(const bf16x8*)(r + half * 8),
                                *(const bf16x8*)(r + 16 + half * 8));
        }
        #pragma unroll
        for (int nt = 0; nt < 2; ++nt) {
            const __bf16* r = cB + (wn * 32 + nt * 16 + l16) * LDSW;
            bfr[nt] = make_frag(*(const bf16x8*)(r + half * 8),
                                *(const bf16x8*)(r + 16 + half * 8));
        }
        #pragma unroll
        for (int mt = 0; mt < 4; ++mt)
            #pragma unroll
            for (int nt = 0; nt < 2; ++nt)
                acc[mt][nt] = __builtin_amdgcn_wmma_f32_16x16x32_bf16(
                    false, afr[mt], false, bfr[nt], (short)0, acc[mt][nt], false, false);
        __syncthreads();
    }

    // epilogue: lane element (r) -> m_local = r + 8*half, n_local = l16
    #pragma unroll
    for (int nt = 0; nt < 2; ++nt) {
        int n = bn + wn * 32 + nt * 16 + l16;
        float bv = bias[n];
        #pragma unroll
        for (int mt = 0; mt < 4; ++mt) {
            #pragma unroll
            for (int r = 0; r < 8; ++r) {
                int m = bm + wm * 64 + mt * 16 + half * 8 + r;
                float v = acc[mt][nt][r] + bv;
                if (MODE == 0) {
                    int b = m >> 11, s = m & (S_ - 1);
                    int h = n >> 6,  e = n & 63;
                    ((__bf16*)out)[(((size_t)b * H_ + h) * S_ + s) * HD_ + e] = (__bf16)v;
                } else if (MODE == 1) {
                    int b = m >> 11, s = m & (S_ - 1);
                    int h = n >> 6,  e = n & 63;
                    ((__bf16*)out)[(((size_t)b * H_ + h) * HD_ + e) * S_ + s] = (__bf16)v;
                } else if (MODE == 2) {
                    size_t idx = (size_t)m * N + n;
                    ((float*)out)[idx] = v + resid[idx];
                } else {
                    ((__bf16*)out)[(size_t)m * N + n] = (__bf16)fmaxf(v, 0.f);
                }
            }
        }
    }
}

// ---------------------------------------------------------------------------
// Flash attention: per block one (b,h) and 64 query rows; 4 waves, each owns
// 16 query rows. KV tiles of 128 keys. Q*K^T and P*V via WMMA; online softmax.
// K read direct from [B,H,S,HD] (natural Bt layout); V pre-transposed [B,H,HD,S].
// ---------------------------------------------------------------------------
__global__ __launch_bounds__(128) void attn_flash(
    const __bf16* __restrict__ Qm, const __bf16* __restrict__ Km,
    const __bf16* __restrict__ Vtm, __bf16* __restrict__ Oc)
{
    const int bh   = blockIdx.y;
    const int b    = bh / H_, h = bh % H_;
    const int q0   = blockIdx.x * 64;
    const int tid  = threadIdx.x;
    const int lane = tid & 31;
    const int wave = tid >> 5;
    const int half = lane >> 4;
    const int l16  = lane & 15;

    const size_t base = (size_t)bh * S_ * HD_;
    const __bf16* qp = Qm  + base;
    const __bf16* kp = Km  + base;
    const __bf16* vp = Vtm + base;

    __shared__ __align__(16) __bf16 Pl[4][16][136];  // per-wave P staging

    // Q fragments: 2 K-chunks over HD=64, rows = q0 + wave*16 + l16
    bf16x16 qf[2];
    {
        const __bf16* qr = qp + (size_t)(q0 + wave * 16 + l16) * HD_;
        #pragma unroll
        for (int c = 0; c < 2; ++c)
            qf[c] = make_frag(*(const bf16x8*)(qr + c * 32 + half * 8),
                              *(const bf16x8*)(qr + c * 32 + 16 + half * 8));
    }

    f32x8 ot[4];
    for (int i = 0; i < 4; ++i) ot[i] = zero8();
    float st_m[8], st_l[8];
    for (int r = 0; r < 8; ++r) { st_m[r] = -3.0e38f; st_l[r] = 0.f; }

    for (int kv0 = 0; kv0 < S_; kv0 += 128) {
        // --- scores: S = Q * K^T / 8 ---
        f32x8 sc[8];
        #pragma unroll
        for (int nt = 0; nt < 8; ++nt) {
            f32x8 a = zero8();
            #pragma unroll
            for (int c = 0; c < 2; ++c) {
                const __bf16* kr = kp + (size_t)(kv0 + nt * 16 + l16) * HD_ + c * 32 + half * 8;
                bf16x16 kf = make_frag(*(const bf16x8*)kr, *(const bf16x8*)(kr + 16));
                a = __builtin_amdgcn_wmma_f32_16x16x32_bf16(
                        false, qf[c], false, kf, (short)0, a, false, false);
            }
            sc[nt] = a * 0.125f;
        }
        // --- online softmax stats (rows shared by the 16 lanes of a half) ---
        float corr[8], sl[8];
        #pragma unroll
        for (int r = 0; r < 8; ++r) {
            float mv = sc[0][r];
            #pragma unroll
            for (int nt = 1; nt < 8; ++nt) mv = fmaxf(mv, sc[nt][r]);
            for (int off = 8; off >= 1; off >>= 1)
                mv = fmaxf(mv, __shfl_xor(mv, off, 32));
            float mn = fmaxf(st_m[r], mv);
            corr[r] = __expf(st_m[r] - mn);
            st_m[r] = mn;
            sl[r] = 0.f;
        }
        #pragma unroll
        for (int nt = 0; nt < 8; ++nt) {
            #pragma unroll
            for (int r = 0; r < 8; ++r) {
                float p = __expf(sc[nt][r] - st_m[r]);
                sl[r] += p;
                Pl[wave][half * 8 + r][nt * 16 + l16] = (__bf16)p;
            }
        }
        #pragma unroll
        for (int r = 0; r < 8; ++r) {
            float s = sl[r];
            for (int off = 8; off >= 1; off >>= 1) s += __shfl_xor(s, off, 32);
            st_l[r] = st_l[r] * corr[r] + s;
        }
        #pragma unroll
        for (int nt = 0; nt < 4; ++nt)
            #pragma unroll
            for (int r = 0; r < 8; ++r) ot[nt][r] *= corr[r];
        __syncthreads();

        // --- O += P * V ---
        #pragma unroll
        for (int c = 0; c < 4; ++c) {
            const __bf16* pr = &Pl[wave][l16][0];
            bf16x16 pf = make_frag(*(const bf16x8*)(pr + c * 32 + half * 8),
                                   *(const bf16x8*)(pr + c * 32 + 16 + half * 8));
            #pragma unroll
            for (int nt = 0; nt < 4; ++nt) {
                const __bf16* vr = vp + (size_t)(nt * 16 + l16) * S_ + kv0 + c * 32 + half * 8;
                bf16x16 vf = make_frag(*(const bf16x8*)vr, *(const bf16x8*)(vr + 16));
                ot[nt] = __builtin_amdgcn_wmma_f32_16x16x32_bf16(
                            false, pf, false, vf, (short)0, ot[nt], false, false);
            }
        }
        __syncthreads();
    }

    // write concat [B, S, H*HD] as bf16 (A-operand of proj GEMM)
    #pragma unroll
    for (int nt = 0; nt < 4; ++nt) {
        #pragma unroll
        for (int r = 0; r < 8; ++r) {
            int s = q0 + wave * 16 + half * 8 + r;
            int e = nt * 16 + l16;
            float v = ot[nt][r] / st_l[r];
            Oc[((size_t)b * S_ + s) * D_ + h * HD_ + e] = (__bf16)v;
        }
    }
}

// ---------------------------------------------------------------------------
// LayerNorm over D=1024 per row; fp32 out (+ optional bf16 copy for next GEMM)
// ---------------------------------------------------------------------------
__global__ __launch_bounds__(256) void layernorm_k(
    const float* __restrict__ in, const float* __restrict__ gamma,
    const float* __restrict__ beta, float* __restrict__ outf,
    __bf16* __restrict__ outb, int emit_bf16)
{
    const int row = blockIdx.x;
    const int tid = threadIdx.x;
    const float* x = in + (size_t)row * D_;

    float v[4], s = 0.f, ss = 0.f;
    #pragma unroll
    for (int i = 0; i < 4; ++i) {
        v[i] = x[tid + i * 256];
        s += v[i]; ss += v[i] * v[i];
    }
    for (int off = 16; off >= 1; off >>= 1) {
        s  += __shfl_xor(s,  off, 32);
        ss += __shfl_xor(ss, off, 32);
    }
    __shared__ float red[2][8];
    int wv = tid >> 5, ln = tid & 31;
    if (ln == 0) { red[0][wv] = s; red[1][wv] = ss; }
    __syncthreads();
    s = 0.f; ss = 0.f;
    #pragma unroll
    for (int w = 0; w < 8; ++w) { s += red[0][w]; ss += red[1][w]; }

    float mean = s * (1.f / D_);
    float var  = ss * (1.f / D_) - mean * mean;
    float inv  = rsqrtf(var + 1e-10f);
    #pragma unroll
    for (int i = 0; i < 4; ++i) {
        int c = tid + i * 256;
        float y = (v[i] - mean) * inv * gamma[c] + beta[c];
        outf[(size_t)row * D_ + c] = y;
        if (emit_bf16) outb[(size_t)row * D_ + c] = (__bf16)y;
    }
}

// ---------------------------------------------------------------------------
extern "C" void kernel_launch(void* const* d_in, const int* in_sizes, int n_in,
                              void* d_out, int out_size, void* d_ws, size_t ws_size,
                              hipStream_t stream) {
    const float* x   = (const float*)d_in[0];
    const float* wq  = (const float*)d_in[1];
    const float* bq  = (const float*)d_in[2];
    const float* wk  = (const float*)d_in[3];
    const float* bk  = (const float*)d_in[4];
    const float* wv  = (const float*)d_in[5];
    const float* bv  = (const float*)d_in[6];
    const float* wp  = (const float*)d_in[7];
    const float* bp  = (const float*)d_in[8];
    const float* g1  = (const float*)d_in[9];
    const float* be1 = (const float*)d_in[10];
    const float* w1  = (const float*)d_in[11];
    const float* b1  = (const float*)d_in[12];
    const float* w2  = (const float*)d_in[13];
    const float* b2  = (const float*)d_in[14];
    const float* g2  = (const float*)d_in[15];
    const float* be2 = (const float*)d_in[16];
    float* out = (float*)d_out;

    char* ws = (char*)d_ws;
    const size_t MB = 1024 * 1024;
    __bf16* XB    = (__bf16*)(ws + 0 * MB);    // x bf16          [4096][1024]   8MB
    __bf16* WTQ   = (__bf16*)(ws + 8 * MB);    // wq^T            [1024][1024]   2MB
    __bf16* WTK   = (__bf16*)(ws + 10 * MB);
    __bf16* WTV   = (__bf16*)(ws + 12 * MB);
    __bf16* WTP   = (__bf16*)(ws + 14 * MB);   // w_proj^T                       2MB
    __bf16* WT1   = (__bf16*)(ws + 16 * MB);   // w1^T            [4096][1024]   8MB
    __bf16* WT2   = (__bf16*)(ws + 24 * MB);   // w2^T            [1024][4096]   8MB
    __bf16* QB    = (__bf16*)(ws + 32 * MB);   // Q [B,H,S,HD]                   8MB
    __bf16* KB    = (__bf16*)(ws + 40 * MB);   // K [B,H,S,HD]                   8MB
    __bf16* VT    = (__bf16*)(ws + 48 * MB);   // V^T [B,H,HD,S]                 8MB
    __bf16* CC    = (__bf16*)(ws + 56 * MB);   // concat bf16     [4096][1024]   8MB
    float*  RES   = (float*)(ws + 64 * MB);    // residual fp32   [4096][1024]  16MB
    float*  OUT1F = (float*)(ws + 80 * MB);    // out1 fp32                     16MB
    __bf16* OUT1B = (__bf16*)(ws + 96 * MB);   // out1 bf16                      8MB
    __bf16* H1    = (__bf16*)(ws + 104 * MB);  // relu(ffn1)      [4096][4096]  32MB

    dim3 blk(256);
    cvt_f32_bf16<<<(MSZ * D_) / 256, blk, 0, stream>>>(x, XB, MSZ * D_);
    transpose_wqkv<<<(H_ * D_ * HD_) / 256, blk, 0, stream>>>(wq, WTQ);
    transpose_wqkv<<<(H_ * D_ * HD_) / 256, blk, 0, stream>>>(wk, WTK);
    transpose_wqkv<<<(H_ * D_ * HD_) / 256, blk, 0, stream>>>(wv, WTV);
    transpose_wt<<<(D_ * D_) / 256, blk, 0, stream>>>(wp, WTP, D_, D_);
    transpose_wt<<<(D_ * FFN_) / 256, blk, 0, stream>>>(w1, WT1, D_, FFN_);
    transpose_wt<<<(FFN_ * D_) / 256, blk, 0, stream>>>(w2, WT2, FFN_, D_);

    // QKV projections
    gemm_bf16<0><<<dim3(D_ / 128, MSZ / 128), blk, 0, stream>>>(XB, WTQ, bq, nullptr, QB, MSZ, D_, D_);
    gemm_bf16<0><<<dim3(D_ / 128, MSZ / 128), blk, 0, stream>>>(XB, WTK, bk, nullptr, KB, MSZ, D_, D_);
    gemm_bf16<1><<<dim3(D_ / 128, MSZ / 128), blk, 0, stream>>>(XB, WTV, bv, nullptr, VT, MSZ, D_, D_);

    // flash attention
    attn_flash<<<dim3(S_ / 64, B_ * H_), dim3(128), 0, stream>>>(QB, KB, VT, CC);

    // proj + residual(x) -> LN1
    gemm_bf16<2><<<dim3(D_ / 128, MSZ / 128), blk, 0, stream>>>(CC, WTP, bp, x, RES, MSZ, D_, D_);
    layernorm_k<<<MSZ, blk, 0, stream>>>(RES, g1, be1, OUT1F, OUT1B, 1);

    // FFN
    gemm_bf16<3><<<dim3(FFN_ / 128, MSZ / 128), blk, 0, stream>>>(OUT1B, WT1, b1, nullptr, H1, MSZ, FFN_, D_);
    gemm_bf16<2><<<dim3(D_ / 128, MSZ / 128), blk, 0, stream>>>(H1, WT2, b2, OUT1F, RES, MSZ, D_, FFN_);
    layernorm_k<<<MSZ, blk, 0, stream>>>(RES, g2, be2, out, nullptr, 0);
}
